// WindowAttention_35081292874052
// MI455X (gfx1250) — compile-verified, matrix operand order checked by reading, not run on hardware
//
#include <hip/hip_runtime.h>

typedef __attribute__((ext_vector_type(16))) _Float16 v16h;
typedef __attribute__((ext_vector_type(8)))  _Float16 v8h;
typedef __attribute__((ext_vector_type(8)))  float    v8f;

#define DIM   192
#define NH    6
#define HD    32
#define NTOK  64
#define IMGH  224
#define IMGW  224
#define NWIN  28
#define SHIFT 4
#define ATTN_SCALE 0.17677669529663687f  // 1/sqrt(32)

// LDS layout (in halves):
//   [0,     12288)  : window tile 64x192 (phase 0/1), reused as O 64x192 (phase 2/3)
//   [12288, 49152)  : q [NH][64][32], k [NH][64][32], v TRANSPOSED [NH][32][64]
//   [49152, 73728)  : P as [NH][64][64] f16
#define OFF_WIN 0
#define OFF_QKV 12288
#define OFF_P   49152
#define LDS_HALVES 73728

// Workspace layout:
//   [0,      221184) : qkv_w as f16 (576*192)
//   [221184, 294912) : proj_w as f16 (192*192)
//   [294912, 393216) : bias table f32 [NH][64][64]
#define WS_PROJW_OFF  (576 * 192)              // in halves
#define WS_BIAS_OFF   (576 * 192 + 192 * 192)  // halves before bias region
#define WS_BYTES      (2u * (576u * 192u + 192u * 192u) + 4u * NH * NTOK * NTOK)

// ---- WMMA helpers (A: 16x32 MxK row tiles; B: 32x16 KxN, column n = one source row) ----

static __device__ inline v8f wmma32(v16h a, v16h b, v8f c) {
  return __builtin_amdgcn_wmma_f32_16x16x32_f16(false, a, false, b, (short)0, c, false, false);
}

// A-tile (or row-major B-tile) from f16 memory (LDS or global):
// lane holds row m0+(lane&15); halves 0..7 = K kb..kb+7, 8..15 = K kb+16..kb+23,
// kb = k0 + (lane>=16)*8
static __device__ inline v16h tile_rowmajor_f16(const _Float16* base, int m0, int ld, int k0, int lane) {
  int m  = m0 + (lane & 15);
  int kb = k0 + ((lane >> 4) & 1) * 8;
  const _Float16* p = base + m * ld + kb;
  v8h lo = *(const v8h*)(p);
  v8h hi = *(const v8h*)(p + 16);
  return __builtin_shufflevector(lo, hi, 0,1,2,3,4,5,6,7,8,9,10,11,12,13,14,15);
}

// B-tile whose column n is a row of a row-major f32 global matrix (fallback path)
static __device__ inline v16h tile_rowmajor_f32(const float* src, int n0, int ld, int k0, int lane) {
  int n  = n0 + (lane & 15);
  int kb = k0 + ((lane >> 4) & 1) * 8;
  const float* p = src + n * ld + kb;
  float4 a0 = *(const float4*)(p);
  float4 a1 = *(const float4*)(p + 4);
  float4 a2 = *(const float4*)(p + 16);
  float4 a3 = *(const float4*)(p + 20);
  v16h b;
  b[0]=(_Float16)a0.x;  b[1]=(_Float16)a0.y;  b[2]=(_Float16)a0.z;  b[3]=(_Float16)a0.w;
  b[4]=(_Float16)a1.x;  b[5]=(_Float16)a1.y;  b[6]=(_Float16)a1.z;  b[7]=(_Float16)a1.w;
  b[8]=(_Float16)a2.x;  b[9]=(_Float16)a2.y;  b[10]=(_Float16)a2.z; b[11]=(_Float16)a2.w;
  b[12]=(_Float16)a3.x; b[13]=(_Float16)a3.y; b[14]=(_Float16)a3.z; b[15]=(_Float16)a3.w;
  return b;
}

// ---------------- Prep: convert weights to f16, expand rel-pos bias ----------------
__global__ __launch_bounds__(256)
void swin_prep_kernel(const float* __restrict__ qkv_w,
                      const float* __restrict__ proj_w,
                      const float* __restrict__ rel_pos_bias,
                      const int*   __restrict__ rel_pos_idx,
                      _Float16* __restrict__ qkv_w_h,
                      _Float16* __restrict__ proj_w_h,
                      float*    __restrict__ bias_tab) {
  int i = blockIdx.x * blockDim.x + threadIdx.x;
  int stride = gridDim.x * blockDim.x;
  for (int e = i; e < 576 * DIM; e += stride) qkv_w_h[e]  = (_Float16)qkv_w[e];
  for (int e = i; e < DIM * DIM; e += stride) proj_w_h[e] = (_Float16)proj_w[e];
  for (int e = i; e < NTOK * NTOK; e += stride) {
    int idx = rel_pos_idx[e];
#pragma unroll
    for (int h = 0; h < NH; ++h)
      bias_tab[h * (NTOK * NTOK) + e] = rel_pos_bias[idx * NH + h];
  }
}

// ---------------- Fused per-window attention ----------------
template <bool USE_WS>
__global__ __launch_bounds__(256)
void swin_window_attn_kernel(const float* __restrict__ x,
                             const float* __restrict__ qkv_w,
                             const float* __restrict__ qkv_b,
                             const float* __restrict__ proj_w,
                             const float* __restrict__ proj_b,
                             const float* __restrict__ rel_pos_bias,
                             const int*   __restrict__ rel_pos_idx,
                             const _Float16* __restrict__ qkv_w_h,
                             const _Float16* __restrict__ proj_w_h,
                             const float*    __restrict__ bias_tab,
                             float* __restrict__ out) {
  extern __shared__ _Float16 smem[];
  _Float16* winb = smem + OFF_WIN;   // 64 x 192 (phase0/1), O buffer (phase2/3)
  _Float16* qkvb = smem + OFF_QKV;   // q,k row-major; v transposed
  _Float16* pbuf = smem + OFF_P;     // [NH][64][64]

  const int wid  = blockIdx.x;
  const int bimg = wid / (NWIN * NWIN);
  const int rem  = wid % (NWIN * NWIN);
  const int wh   = rem / NWIN;
  const int ww   = rem % NWIN;
  const int hb   = wh * 8 + SHIFT;   // 4..220, +7 max 227 < 2*224
  const int wb   = ww * 8 + SHIFT;

  const int tid  = threadIdx.x;
  const int lane = tid & 31;
  // wave index is uniform across the wave: force it scalar so tile loops /
  // q-k-v store selection compile to SALU control flow, not exec-mask dances.
  const int wave = __builtin_amdgcn_readfirstlane(tid >> 5);
  const int ncol = lane & 15;
  const int hi8  = ((lane >> 4) & 1) * 8;

  // ---------------- Phase 0: gather rolled window into LDS (f16) ----------------
  // x is streamed once -> non-temporal loads keep L2 for weights/bias.
  for (int e = tid; e < NTOK * DIM; e += 256) {
    int c   = e >> 6;        // channel
    int tok = e & 63;        // token (fast index -> contiguous w reads)
    int h = hb + (tok >> 3); if (h >= IMGH) h -= IMGH;   // roll = cond-subtract
    int w = wb + (tok & 7);  if (w >= IMGW) w -= IMGW;
    float xv = __builtin_nontemporal_load(&x[((bimg * DIM + c) * IMGH + h) * IMGW + w]);
    winb[tok * DIM + c] = (_Float16)xv;
  }
  __syncthreads();

  // ---------------- Phase 1: qkv = win @ Wqkv^T + b  (64x192 @ 192x576) ----------------
  // mt-outer: load A tiles once, reuse across this wave's nt tiles.
  for (int mt = 0; mt < 4; ++mt) {
    v16h a[6];
#pragma unroll
    for (int ks = 0; ks < 6; ++ks)
      a[ks] = tile_rowmajor_f16(winb, mt * 16, DIM, ks * 32, lane);
    for (int nt = wave; nt < 36; nt += 8) {
      v8f acc = {};
#pragma unroll
      for (int ks = 0; ks < 6; ++ks) {
        v16h b = USE_WS ? tile_rowmajor_f16(qkv_w_h, nt * 16, DIM, ks * 32, lane)
                        : tile_rowmajor_f32(qkv_w, nt * 16, DIM, ks * 32, lane);
        acc = wmma32(a[ks], b, acc);
      }
      int n  = nt * 16 + ncol;           // 0..575
      float bv = qkv_b[n];
      int m0 = mt * 16 + hi8;
      if (nt < 24) {
        // q / k: row-major [part][head][tok][dim] -> scattered b16 stores
        int part = (nt >= 12) ? 1 : 0;   // scalar: nt is SGPR
        int nn   = n - part * DIM;
        int head = nn >> 5, dm = nn & 31;
        _Float16* dst = qkvb + part * (NH * NTOK * HD) + head * (NTOK * HD) + dm;
#pragma unroll
        for (int r = 0; r < 8; ++r)
          dst[(m0 + r) * HD] = (_Float16)(acc[r] + bv);
      } else {
        // v: TRANSPOSED [head][dim][tok] -> lane's 8 M-values are contiguous: one b128
        int nn   = n - 2 * DIM;
        int head = nn >> 5, dm = nn & 31;
        _Float16* dst = qkvb + 2 * (NH * NTOK * HD) + head * (HD * NTOK) + dm * NTOK + m0;
        v8h pk;
#pragma unroll
        for (int r = 0; r < 8; ++r) pk[r] = (_Float16)(acc[r] + bv);
        *(v8h*)dst = pk;
      }
    }
  }
  __syncthreads();

  // ---------------- Phase 2: per-head attention (waves 0..5) ----------------
  if (wave < NH) {
    const int head = wave;
    const _Float16* q  = qkvb + head * (NTOK * HD);
    const _Float16* k  = qkvb + (NH * NTOK * HD) + head * (NTOK * HD);
    const _Float16* vt_base = qkvb + 2 * (NH * NTOK * HD) + head * (HD * NTOK); // [32][64]
    _Float16* p = pbuf + head * (NTOK * NTOK);

    // S = q @ k^T : 4x4 tiles, single K=32 step each; hoist the 4 K-tiles.
    v16h kt[4];
#pragma unroll
    for (int tj = 0; tj < 4; ++tj)
      kt[tj] = tile_rowmajor_f16(k, tj * 16, HD, 0, lane);

    v8f s[4][4];
#pragma unroll
    for (int ti = 0; ti < 4; ++ti) {
      v16h a = tile_rowmajor_f16(q, ti * 16, HD, 0, lane);
#pragma unroll
      for (int tj = 0; tj < 4; ++tj) {
        v8f c = {};
        s[ti][tj] = wmma32(a, kt[tj], c);
      }
    }

    // scale + relative position bias
#pragma unroll
    for (int ti = 0; ti < 4; ++ti)
#pragma unroll
      for (int tj = 0; tj < 4; ++tj)
#pragma unroll
        for (int r = 0; r < 8; ++r) {
          int m = ti * 16 + hi8 + r;
          int n = tj * 16 + ncol;
          float bval = USE_WS ? bias_tab[head * (NTOK * NTOK) + m * NTOK + n]
                              : rel_pos_bias[rel_pos_idx[m * NTOK + n] * NH + head];
          s[ti][tj][r] = s[ti][tj][r] * ATTN_SCALE + bval;
        }

    // softmax per row (row lives on one 16-lane half across 4 tj-registers)
#pragma unroll
    for (int ti = 0; ti < 4; ++ti) {
#pragma unroll
      for (int r = 0; r < 8; ++r) {
        float mx = -3.0e38f;
#pragma unroll
        for (int tj = 0; tj < 4; ++tj) mx = fmaxf(mx, s[ti][tj][r]);
#pragma unroll
        for (int o = 1; o < 16; o <<= 1) mx = fmaxf(mx, __shfl_xor(mx, o, 32));
        float sum = 0.0f;
#pragma unroll
        for (int tj = 0; tj < 4; ++tj) {
          float e = __expf(s[ti][tj][r] - mx);
          s[ti][tj][r] = e;
          sum += e;
        }
#pragma unroll
        for (int o = 1; o < 16; o <<= 1) sum += __shfl_xor(sum, o, 32);
        float rinv = 1.0f / sum;
        int m = ti * 16 + hi8 + r;
#pragma unroll
        for (int tj = 0; tj < 4; ++tj)
          p[m * NTOK + tj * 16 + ncol] = (_Float16)(s[ti][tj][r] * rinv);
      }
    }

    // O = P @ V : 4x2 tiles, K = 64 (2 steps).
    // V is transposed in LDS -> B-tile column n is row n of vt: vectorized loads.
    v16h vtile[2][2];
#pragma unroll
    for (int nt = 0; nt < 2; ++nt)
#pragma unroll
      for (int ks = 0; ks < 2; ++ks)
        vtile[nt][ks] = tile_rowmajor_f16(vt_base, nt * 16, NTOK, ks * 32, lane);

#pragma unroll
    for (int ti = 0; ti < 4; ++ti) {
#pragma unroll
      for (int nt = 0; nt < 2; ++nt) {
        v8f acc = {};
#pragma unroll
        for (int ks = 0; ks < 2; ++ks) {
          v16h a = tile_rowmajor_f16(p, ti * 16, NTOK, ks * 32, lane);
          acc = wmma32(a, vtile[nt][ks], acc);
        }
        int m0 = ti * 16 + hi8;
#pragma unroll
        for (int r = 0; r < 8; ++r)
          winb[(m0 + r) * DIM + head * HD + nt * 16 + ncol] = (_Float16)acc[r];
      }
    }
  } else {
    // idle waves: warm proj weights into cache for phase 3
    if (USE_WS) {
      for (int r = (wave - NH) * 32 + lane; r < DIM; r += 64)
        __builtin_prefetch(proj_w_h + r * DIM, 0, 1);
    } else {
      for (int r = (wave - NH) * 32 + lane; r < DIM; r += 64)
        __builtin_prefetch(proj_w + r * DIM, 0, 1);
    }
  }
  __syncthreads();

  // ---------------- Phase 3: out = O @ Wproj^T + b, scatter with roll-back ----------------
  for (int mt = 0; mt < 4; ++mt) {
    v16h a[6];
#pragma unroll
    for (int ks = 0; ks < 6; ++ks)
      a[ks] = tile_rowmajor_f16(winb, mt * 16, DIM, ks * 32, lane);

    // per-token output offsets depend only on (mt, lane, r): hoist out of nt loop
    int hw_off[8];
#pragma unroll
    for (int r = 0; r < 8; ++r) {
      int tok = mt * 16 + hi8 + r;
      int h = hb + (tok >> 3); if (h >= IMGH) h -= IMGH;
      int w = wb + (tok & 7);  if (w >= IMGW) w -= IMGW;
      hw_off[r] = h * IMGW + w;
    }

    for (int nt = wave; nt < 12; nt += 8) {
      v8f acc = {};
#pragma unroll
      for (int ks = 0; ks < 6; ++ks) {
        v16h b = USE_WS ? tile_rowmajor_f16(proj_w_h, nt * 16, DIM, ks * 32, lane)
                        : tile_rowmajor_f32(proj_w, nt * 16, DIM, ks * 32, lane);
        acc = wmma32(a[ks], b, acc);
      }
      int c  = nt * 16 + ncol;
      float bv = proj_b[c];
      float* outc = out + (size_t)(bimg * DIM + c) * (IMGH * IMGW);
#pragma unroll
      for (int r = 0; r < 8; ++r)
        __builtin_nontemporal_store(acc[r] + bv, &outc[hw_off[r]]);
    }
  }
}

extern "C" void kernel_launch(void* const* d_in, const int* in_sizes, int n_in,
                              void* d_out, int out_size, void* d_ws, size_t ws_size,
                              hipStream_t stream) {
  const float* x            = (const float*)d_in[0];
  const float* qkv_w        = (const float*)d_in[1];
  const float* qkv_b        = (const float*)d_in[2];
  const float* proj_w       = (const float*)d_in[3];
  const float* proj_b       = (const float*)d_in[4];
  const float* rel_pos_bias = (const float*)d_in[5];
  const int*   rel_pos_idx  = (const int*)d_in[6];
  float* out = (float*)d_out;

  int Bsz = in_sizes[0] / (DIM * IMGH * IMGW);
  size_t shmem = (size_t)LDS_HALVES * 2u;  // 147456 bytes (< 320KB per WGP)
  dim3 grid(Bsz * NWIN * NWIN);

  _Float16* qkv_w_h  = (_Float16*)d_ws;
  _Float16* proj_w_h = qkv_w_h + WS_PROJW_OFF;
  float*    bias_tab = (float*)((char*)d_ws + 2u * (size_t)WS_BIAS_OFF);

  if (ws_size >= (size_t)WS_BYTES) {
    (void)hipFuncSetAttribute((const void*)swin_window_attn_kernel<true>,
                              hipFuncAttributeMaxDynamicSharedMemorySize, (int)shmem);
    swin_prep_kernel<<<96, 256, 0, stream>>>(qkv_w, proj_w, rel_pos_bias, rel_pos_idx,
                                             qkv_w_h, proj_w_h, bias_tab);
    swin_window_attn_kernel<true><<<grid, 256, shmem, stream>>>(
        x, qkv_w, qkv_b, proj_w, proj_b, rel_pos_bias, rel_pos_idx,
        qkv_w_h, proj_w_h, bias_tab, out);
  } else {
    (void)hipFuncSetAttribute((const void*)swin_window_attn_kernel<false>,
                              hipFuncAttributeMaxDynamicSharedMemorySize, (int)shmem);
    swin_window_attn_kernel<false><<<grid, 256, shmem, stream>>>(
        x, qkv_w, qkv_b, proj_w, proj_b, rel_pos_bias, rel_pos_idx,
        qkv_w_h, proj_w_h, bias_tab, out);
  }
}
